// TurboQuantPipeline_2430951490010
// MI455X (gfx1250) — compile-verified, matrix operand order checked by reading, not run on hardware
//
#include <hip/hip_runtime.h>
#include <hip/hip_bf16.h>
#include <math.h>

typedef __attribute__((ext_vector_type(16))) _Float16 v16h;
typedef __attribute__((ext_vector_type(8)))  _Float16 v8h;
typedef __attribute__((ext_vector_type(8)))  float    v8f;

// Problem constants (from reference setup_inputs)
#define HN   32        // heads
#define TN   8192      // sequence length
#define KQ   16        // query rows
#define DD   128       // head dim
#define BT   128       // T-block per workgroup
#define NB   (TN / BT) // 64 blocks per head
#define LDK  (DD + 8)  // padded LDS stride (halves) -> 272B rows, 16B aligned
#define LDT  (BT + 8)
#define PARTIAL_STRIDE 2080  // 16*128 acc + 16 m + 16 l floats

// --- 3-bit polar quantization tables (LEVELS=8) ---
// Standard dims: theta = acos(c) in [0,pi], idx = round(theta/pi*7)
// => idx = #{ c < cos((m+0.5)*pi/7) }, m=0..6
__constant__ float BND_STD[7] = { 0.97492791f, 0.78183148f, 0.43388374f, 0.0f,
                                 -0.43388374f,-0.78183148f,-0.97492791f };
__constant__ float SIN_STD[8] = { 0.0f, 0.43388374f, 0.78183148f, 0.97492791f,
                                  0.97492791f, 0.78183148f, 0.43388374f, 0.0f };
__constant__ float COS_STD[8] = { 1.0f, 0.90096887f, 0.62348980f, 0.22252093f,
                                 -0.22252093f,-0.62348980f,-0.90096887f,-1.0f };
// Last dim: range 2*pi. a = #{ c < cos((m+0.5)*2pi/7) }, m=0..2 ; idx = xlast<0 ? 7-a : a
__constant__ float BND_LST[3] = { 0.90096887f, 0.22252093f, -0.62348980f };
__constant__ float SIN_LST[8] = { 0.0f, 0.78183148f, 0.97492791f, 0.43388374f,
                                 -0.43388374f,-0.97492791f,-0.78183148f, 0.0f };
__constant__ float COS_LST[8] = { 1.0f, 0.62348980f,-0.22252093f,-0.90096887f,
                                 -0.90096887f,-0.22252093f, 0.62348980f, 1.0f };

__device__ __forceinline__ v8f wmma16(v16h a, v16h b, v8f c) {
  return __builtin_amdgcn_wmma_f32_16x16x32_f16(false, a, false, b, (short)0, c,
                                                false, false);
}

__device__ __forceinline__ v16h ld16(const _Float16* lo, const _Float16* hi) {
  v8h a = *(const v8h*)lo;
  v8h b = *(const v8h*)hi;
  v16h r;
#pragma unroll
  for (int i = 0; i < 8; ++i) { r[i] = a[i]; r[i + 8] = b[i]; }
  return r;
}

// One wave processes one 128-float row: quantize->dequantize roundtrip.
// lane holds elements [4*lane, 4*lane+3]. Result in xh.
__device__ __forceinline__ void polar_roundtrip_row(const float* __restrict__ src,
                                                    int lane, float4& xh) {
  float4 x = ((const float4*)src)[lane];
  float q0 = x.x * x.x, q1 = x.y * x.y, q2 = x.z * x.z, q3 = x.w * x.w;
  // within-lane suffix sums of squares
  float s3 = q3, s2 = q3 + q2, s1 = s2 + q1, s0 = s1 + q0;
  // wave inclusive suffix scan of per-lane totals
  float S = s0;
#pragma unroll
  for (int off = 1; off < 32; off <<= 1) {
    float v = __shfl_down(S, off, 32);
    if (lane + off < 32) S += v;
  }
  float total = __shfl(S, 0, 32);
  float r = sqrtf(total);
  float hi = S - s0;  // suffix from higher lanes
  float es[4] = { s0 + hi, s1 + hi, s2 + hi, s3 + hi };
  float xs[4] = { x.x, x.y, x.z, x.w };
  float xlast = __shfl(x.w, 31, 32);  // x[127]

  float sn[4], cs[4];
#pragma unroll
  for (int e = 0; e < 4; ++e) {
    int j = lane * 4 + e;
    float n = sqrtf(es[e]);
    float c = xs[e] / fmaxf(n, 1e-12f);
    c = fminf(fmaxf(c, -1.0f), 1.0f);
    if (j < 126) {
      int idx = 0;
#pragma unroll
      for (int m = 0; m < 7; ++m) idx += (c < BND_STD[m]) ? 1 : 0;
      sn[e] = SIN_STD[idx];
      cs[e] = COS_STD[idx];
    } else if (j == 126) {
      int a = 0;
#pragma unroll
      for (int m = 0; m < 3; ++m) a += (c < BND_LST[m]) ? 1 : 0;
      int idx = (xlast < 0.0f) ? (7 - a) : a;
      sn[e] = SIN_LST[idx];
      cs[e] = COS_LST[idx];
    } else {  // j == 127: cos term is 1, sin placeholder 1 (not in products past end)
      sn[e] = 1.0f;
      cs[e] = 1.0f;
    }
  }
  // exclusive prefix product of sines across the row
  float p = sn[0] * sn[1] * sn[2] * sn[3];
  float incl = p;
#pragma unroll
  for (int off = 1; off < 32; off <<= 1) {
    float v = __shfl_up(incl, off, 32);
    if (lane >= off) incl *= v;
  }
  float excl = __shfl_up(incl, 1, 32);
  if (lane == 0) excl = 1.0f;
  float P0 = excl;
  float P1 = P0 * sn[0];
  float P2 = P1 * sn[1];
  float P3 = P2 * sn[2];
  xh.x = r * P0 * cs[0];
  xh.y = r * P1 * cs[1];
  xh.z = r * P2 * cs[2];
  xh.w = r * P3 * cs[3];
}

__global__ __launch_bounds__(256) void attn_partial_kernel(
    const float* __restrict__ Q, const float* __restrict__ Kp,
    const float* __restrict__ Vp, const float* __restrict__ noise,
    float* __restrict__ ws) {
  __shared__ __align__(16) _Float16 Qs[KQ][LDK];
  __shared__ __align__(16) _Float16 Kl[BT][LDK];   // K rows (row-major, d)
  __shared__ __align__(16) _Float16 Vt[DD][LDT];   // V transposed: [d][t]
  __shared__ __align__(16) float    Sb[KQ][LDT];
  __shared__ __align__(16) _Float16 Pb[KQ][LDT];
  __shared__ float mrow[KQ], lrow[KQ];

  const int h = blockIdx.y;
  const int blk = blockIdx.x;
  const int t0 = blk * BT;
  const int tid = threadIdx.x;
  const int lane = tid & 31;
  const int wave = tid >> 5;  // 0..7

  // Q -> LDS f16
  for (int i = tid; i < KQ * DD; i += 256) {
    int m = i >> 7, d = i & 127;
    Qs[m][d] = (_Float16)Q[((size_t)h * KQ + m) * DD + d];
  }
  // Dequantize K and V rows for this T-block (one row per wave per iter)
  for (int rr = 0; rr < BT / 8; ++rr) {
    int t = rr * 8 + wave;
    float4 xh;
    polar_roundtrip_row(Kp + ((size_t)h * TN + t0 + t) * DD, lane, xh);
    Kl[t][lane * 4 + 0] = (_Float16)xh.x;
    Kl[t][lane * 4 + 1] = (_Float16)xh.y;
    Kl[t][lane * 4 + 2] = (_Float16)xh.z;
    Kl[t][lane * 4 + 3] = (_Float16)xh.w;
    polar_roundtrip_row(Vp + ((size_t)h * TN + t0 + t) * DD, lane, xh);
    Vt[lane * 4 + 0][t] = (_Float16)xh.x;
    Vt[lane * 4 + 1][t] = (_Float16)xh.y;
    Vt[lane * 4 + 2][t] = (_Float16)xh.z;
    Vt[lane * 4 + 3][t] = (_Float16)xh.w;
  }
  __syncthreads();

  // S = Q * K^T : each wave owns one 16-wide n-tile (BT/16 == 8 tiles)
  {
    const int tbase = wave * 16;
    const int rsel = lane & 15;
    const int kb = (lane < 16) ? 0 : 8;  // ISA A/B 16-bit operand K split
    v8f c = {};
#pragma unroll
    for (int k0 = 0; k0 < DD; k0 += 32) {
      v16h a = ld16(&Qs[rsel][k0 + kb], &Qs[rsel][k0 + kb + 16]);
      v16h b = ld16(&Kl[tbase + rsel][k0 + kb], &Kl[tbase + rsel][k0 + kb + 16]);
      c = wmma16(a, b, c);
    }
    const int n = lane & 15;
    const int mb = (lane < 16) ? 0 : 8;
#pragma unroll
    for (int i = 0; i < 8; ++i) Sb[mb + i][tbase + n] = c[i];
  }
  __syncthreads();

  // Per-block softmax partial over BT columns: 16 threads per query row
  {
    const int row = tid >> 4;
    const int tcol = tid & 15;
    const float scale = 0.08838834764831845f;  // 1/sqrt(128)
    const float* nrow = noise + ((size_t)h * KQ + row) * TN + t0;
    float vals[8];
    float mymax = -INFINITY;
#pragma unroll
    for (int e = 0; e < 8; ++e) {
      int cidx = tcol + e * 16;
      float v = Sb[row][cidx] * scale + nrow[cidx];
      vals[e] = v;
      mymax = fmaxf(mymax, v);
    }
#pragma unroll
    for (int m = 8; m >= 1; m >>= 1) mymax = fmaxf(mymax, __shfl_xor(mymax, m, 32));
    float lsum = 0.0f;
#pragma unroll
    for (int e = 0; e < 8; ++e) {
      int cidx = tcol + e * 16;
      float p = __expf(vals[e] - mymax);
      Pb[row][cidx] = (_Float16)p;
      lsum += p;
    }
#pragma unroll
    for (int m = 8; m >= 1; m >>= 1) lsum += __shfl_xor(lsum, m, 32);
    if (tcol == 0) { mrow[row] = mymax; lrow[row] = lsum; }
  }
  __syncthreads();

  // acc = P * V_hat : each wave owns one 16-wide d-tile (DD/16 == 8 tiles)
  float* base = ws + ((size_t)h * NB + blk) * PARTIAL_STRIDE;
  {
    const int d0 = wave * 16;
    const int rsel = lane & 15;
    const int kb = (lane < 16) ? 0 : 8;
    v8f c = {};
#pragma unroll
    for (int k0 = 0; k0 < BT; k0 += 32) {
      v16h a = ld16(&Pb[rsel][k0 + kb], &Pb[rsel][k0 + kb + 16]);
      v16h b = ld16(&Vt[d0 + rsel][k0 + kb], &Vt[d0 + rsel][k0 + kb + 16]);
      c = wmma16(a, b, c);
    }
    const int n = lane & 15;
    const int mb = (lane < 16) ? 0 : 8;
#pragma unroll
    for (int i = 0; i < 8; ++i) base[(size_t)(mb + i) * DD + d0 + n] = c[i];
  }
  if (tid < KQ) {
    base[2048 + tid] = mrow[tid];
    base[2064 + tid] = lrow[tid];
  }
}

__global__ __launch_bounds__(256) void combine_kernel(const float* __restrict__ ws,
                                                      float* __restrict__ out) {
  const int h = blockIdx.x;
  const int tid = threadIdx.x;
  const int row = tid >> 4;
  const int c = tid & 15;
  const float* base = ws + (size_t)h * NB * PARTIAL_STRIDE;
  float M = -INFINITY;
  for (int b = 0; b < NB; ++b)
    M = fmaxf(M, base[(size_t)b * PARTIAL_STRIDE + 2048 + row]);
  float L = 0.0f;
  float acc[8];
#pragma unroll
  for (int e = 0; e < 8; ++e) acc[e] = 0.0f;
  for (int b = 0; b < NB; ++b) {
    const float* pb = base + (size_t)b * PARTIAL_STRIDE;
    float w = __expf(pb[2048 + row] - M);
    L += w * pb[2064 + row];
#pragma unroll
    for (int e = 0; e < 8; ++e) {
      int d = c + e * 16;
      acc[e] += w * pb[(size_t)row * DD + d];
    }
  }
  float inv = 1.0f / L;
#pragma unroll
  for (int e = 0; e < 8; ++e) {
    int d = c + e * 16;
    out[((size_t)h * KQ + row) * DD + d] = acc[e] * inv;
  }
}

extern "C" void kernel_launch(void* const* d_in, const int* in_sizes, int n_in,
                              void* d_out, int out_size, void* d_ws, size_t ws_size,
                              hipStream_t stream) {
  const float* Q = (const float*)d_in[0];      // [1,32,16,128]
  const float* K = (const float*)d_in[1];      // [1,32,8192,128]
  const float* V = (const float*)d_in[2];      // [1,32,8192,128]
  const float* noise = (const float*)d_in[3];  // [1,32,16,8192]
  float* out = (float*)d_out;                  // [1,32,16,128]
  float* ws = (float*)d_ws;                    // 32*64*2080 floats (~17 MB)

  dim3 grid(NB, HN);
  attn_partial_kernel<<<grid, 256, 0, stream>>>(Q, K, V, noise, ws);
  combine_kernel<<<HN, 256, 0, stream>>>(ws, out);
}
